// SplitCausalSelfAttention_30537217475190
// MI455X (gfx1250) — matrix operand order repaired and checked by reading, not compiled, not run on hardware
//
#include <hip/hip_runtime.h>
#include <hip/hip_bf16.h>

typedef __attribute__((ext_vector_type(16))) _Float16 v16h;
typedef __attribute__((ext_vector_type(8)))  float    v8f;
typedef __attribute__((ext_vector_type(2)))  _Float16 h2f;
typedef __attribute__((ext_vector_type(4)))  unsigned int u32x4;
typedef __attribute__((ext_vector_type(8)))  int i32x8;
typedef __attribute__((ext_vector_type(4)))  int i32x4;

#define WMMA_F16(a,b,c) __builtin_amdgcn_wmma_f32_16x16x32_f16(false,(a),false,(b),(short)0,(c),false,false)

#define B_DIM 4
#define T_DIM 2048
#define C_DIM 1024
#define H_DIM 16
#define D_DIM 64
#define BT_DIM (B_DIM*T_DIM)

#if defined(__has_builtin)
#  if __has_builtin(__builtin_amdgcn_tensor_load_to_lds)
#    define HAVE_TDM 1
#  else
#    define HAVE_TDM 0
#  endif
#else
#  define HAVE_TDM 0
#endif

// ---------------------------------------------------------------------------
// Tensor Data Mover: 2D tile (strided rows in global -> dense rows in LDS).
// D# packing per CDNA5 ISA 8.3/8.4. data_size = 2 bytes (f16). One call per
// tile from a single wave; completion tracked with TENSORcnt (in-order).
// ---------------------------------------------------------------------------
#if HAVE_TDM
__device__ inline void tdm_load_tile_2d(unsigned ldsByteOff, const _Float16* g,
                                        unsigned tileX, unsigned tileY,
                                        unsigned tensorX, unsigned tensorY,
                                        unsigned strideX /*elems*/) {
  unsigned long long ga = (unsigned long long)(size_t)g;
  u32x4 g0;
  g0[0] = 1u;                                     // count=1, user mode, no gather
  g0[1] = ldsByteOff;                             // lds_addr (bytes)
  g0[2] = (unsigned)(ga & 0xFFFFFFFFu);           // global_addr[31:0]
  g0[3] = (unsigned)((ga >> 32) & 0x1FFFFFFu)     // global_addr[56:32]
        | (2u << 30);                             // type = 2 ("image")
  i32x8 g1;
  g1[0] = (int)(1u << 16);                        // data_size=1 -> 2 bytes
  g1[1] = (int)((tensorX & 0xFFFFu) << 16);       // tensor_dim0[15:0] @ 63:48
  g1[2] = (int)((tensorX >> 16) | ((tensorY & 0xFFFFu) << 16)); // dim0 hi | dim1 lo
  g1[3] = (int)((tensorY >> 16) | (tileX << 16)); // dim1 hi | tile_dim0 @127:112
  g1[4] = (int)(tileY & 0xFFFFu);                 // tile_dim1 @143:128, tile_dim2=0
  g1[5] = (int)strideX;                           // tensor_dim0_stride[31:0]
  g1[6] = 0;                                      // stride0 hi, dim1_stride lo (2D)
  g1[7] = 0;
  i32x4 z4 = {};
  i32x8 z8 = {};
  __builtin_amdgcn_tensor_load_to_lds(g0, g1, z4, z4, z8, 0);
}
#endif

// ---------------------------------------------------------------------------
// Fragment loaders (wave32, 16x16x32 f16 WMMA): contiguous k = 16*(lane/16) + s, same for A and B
// ---------------------------------------------------------------------------
__device__ inline v16h load_fragA(const _Float16* p, int ld) {
  int lane = threadIdx.x & 31;
  const h2f* row = (const h2f*)(p + (size_t)(lane & 15) * ld);
  int hb = (lane >> 4) ? 8 : 0;
  v16h a;
  h2f* ap = (h2f*)&a;
#pragma unroll
  for (int pr = 0; pr < 4; ++pr) {
    ap[pr]     = row[pr + hb];
    ap[pr + 4] = row[pr + 4 + hb];
  }
  return a;
}

// B[k][n] at p[n*ld + k]: lane&15 = col N; lanes 0-15 K=0..15, 16-31 K=16..31.
__device__ inline v16h load_fragB(const _Float16* p, int ld) {
  int lane = threadIdx.x & 31;
  const h2f* col = (const h2f*)(p + (size_t)(lane & 15) * ld + ((lane >> 4) ? 16 : 0));
  v16h b;
  h2f* bp = (h2f*)&b;
#pragma unroll
  for (int pr = 0; pr < 8; ++pr) bp[pr] = col[pr];
  return b;
}

// ---------------------------------------------------------------------------
__global__ void cvt_f32_f16_kernel(const float* __restrict__ src,
                                   _Float16* __restrict__ dst, int n) {
  int i = blockIdx.x * blockDim.x + threadIdx.x;
  if (i < n) dst[i] = (_Float16)src[i];
}

// ---------------------------------------------------------------------------
// GEMM: out[m,n] = sum_k A[m,k] * W[n,k]
// mode 0: f16 -> [B,H,T,D] (Q,K). mode 2: f16 -> [B,H,D,T] (V transposed).
// mode 1: f32 + bias -> [M,N].
// 256 thr = 8 waves (4 M x 2 N); tile 128x64; K slab 32.
// Double-buffered TDM pipeline, K-loop unrolled by 2 so slab bases are
// compile-time constants (keeps ds_load_b128 immediate-offset addressing).
// ---------------------------------------------------------------------------
#define GTM 128
#define GTN 64
#define GKT 32
#define SLAB_HALVES ((GTM + GTN) * GKT)            // 6144 halves = 12288 B
#define SLAB_BYTES  (SLAB_HALVES * 2)
#if HAVE_TDM
#  define GEMM_LDS_BYTES (2 * SLAB_BYTES)          // 24576
#else
#  define GEMM_LDS_BYTES (SLAB_BYTES)
#endif

__device__ inline void gemm_compute_slab(const _Float16* sA, const _Float16* sW,
                                         int wm, int wn, v8f (&acc)[2][2]) {
  v16h a0 = load_fragA(sA + (size_t)(wm * 32) * GKT, GKT);
  v16h a1 = load_fragA(sA + (size_t)(wm * 32 + 16) * GKT, GKT);
  v16h b0 = load_fragB(sW + (size_t)(wn * 32) * GKT, GKT);
  v16h b1 = load_fragB(sW + (size_t)(wn * 32 + 16) * GKT, GKT);
  acc[0][0] = WMMA_F16(a0, b0, acc[0][0]);
  acc[0][1] = WMMA_F16(a0, b1, acc[0][1]);
  acc[1][0] = WMMA_F16(a1, b0, acc[1][0]);
  acc[1][1] = WMMA_F16(a1, b1, acc[1][1]);
}

__global__ __launch_bounds__(256)
void gemm_wmma_kernel(const _Float16* __restrict__ A,
                      const _Float16* __restrict__ W,
                      const float* __restrict__ bias,
                      _Float16* __restrict__ outH,
                      float* __restrict__ outF,
                      int M, int N, int K, int mode) {
  extern __shared__ __align__(16) _Float16 smem[];

  int tid  = threadIdx.x;
  int lane = tid & 31;
  int wave = tid >> 5;
  int wm = wave >> 1;
  int wn = wave & 1;
  int blockM = blockIdx.x * GTM;
  int blockN = blockIdx.y * GTN;

  v8f acc[2][2] = {};

#if HAVE_TDM
  const _Float16* Arow = A + (size_t)blockM * K;
  const _Float16* Wrow = W + (size_t)blockN * K;

  // ---- prologue: DMA slab 0 (k = 0) ----
  if (wave == 0) {
    tdm_load_tile_2d(0, Arow, GKT, GTM, (unsigned)K, (unsigned)M, (unsigned)K);
    tdm_load_tile_2d(GTM * GKT * 2, Wrow, GKT, GTN, (unsigned)K, (unsigned)N, (unsigned)K);
  }
  // K/GKT is even: process 2 slabs per iteration with constant LDS bases.
  for (int k0 = 0; k0 < K; k0 += 2 * GKT) {
    // --- half 1: prefetch slab1 (k0+GKT), compute slab0 (k0) ---
    if (wave == 0) {
      tdm_load_tile_2d(SLAB_BYTES, Arow + k0 + GKT,
                       GKT, GTM, (unsigned)K, (unsigned)M, (unsigned)K);
      tdm_load_tile_2d(SLAB_BYTES + GTM * GKT * 2, Wrow + k0 + GKT,
                       GKT, GTN, (unsigned)K, (unsigned)N, (unsigned)K);
      __builtin_amdgcn_s_wait_tensorcnt(2);   // retire slab0's pair (in-order)
    }
    __syncthreads();
    gemm_compute_slab(smem, smem + GTM * GKT, wm, wn, acc);
    __syncthreads();

    // --- half 2: prefetch slab0 (k0+2*GKT), compute slab1 (k0+GKT) ---
    if (wave == 0) {
      if (k0 + 2 * GKT < K) {
        tdm_load_tile_2d(0, Arow + k0 + 2 * GKT,
                         GKT, GTM, (unsigned)K, (unsigned)M, (unsigned)K);
        tdm_load_tile_2d(GTM * GKT * 2, Wrow + k0 + 2 * GKT,
                         GKT, GTN, (unsigned)K, (unsigned)N, (unsigned)K);
        __builtin_amdgcn_s_wait_tensorcnt(2);
      } else {
        __builtin_amdgcn_s_wait_tensorcnt(0);
      }
    }
    __syncthreads();
    gemm_compute_slab(smem + SLAB_HALVES, smem + SLAB_HALVES + GTM * GKT, wm, wn, acc);
    __syncthreads();
  }
#else
  _Float16* sA = smem;
  _Float16* sW = smem + GTM * GKT;
  for (int k0 = 0; k0 < K; k0 += GKT) {
    {
      int row = tid >> 1, seg = tid & 1;
      const uint4* src = (const uint4*)(A + (size_t)(blockM + row) * K + k0 + seg * 16);
      uint4* dst = (uint4*)(sA + row * GKT + seg * 16);
      dst[0] = src[0];
      dst[1] = src[1];
    }
    {
      int row = tid >> 2, seg = tid & 3;
      const uint4* src = (const uint4*)(W + (size_t)(blockN + row) * K + k0 + seg * 8);
      uint4* dst = (uint4*)(sW + row * GKT + seg * 8);
      dst[0] = src[0];
    }
    __syncthreads();
    gemm_compute_slab(sA, sW, wm, wn, acc);
    __syncthreads();
  }
#endif

  int rlo = (lane >> 4) << 3;
  int cIdx = lane & 15;
#pragma unroll
  for (int fm = 0; fm < 2; ++fm)
#pragma unroll
    for (int fn = 0; fn < 2; ++fn)
#pragma unroll
      for (int r = 0; r < 8; ++r) {
        int m = blockM + wm * 32 + fm * 16 + r + rlo;
        int n = blockN + wn * 32 + fn * 16 + cIdx;
        float v = acc[fm][fn][r];
        if (mode == 0) {                 // [B,H,T,D]
          int b = m >> 11, t = m & (T_DIM - 1);
          int h = n >> 6,  d = n & (D_DIM - 1);
          outH[(((size_t)(b * H_DIM + h) * T_DIM) + t) * D_DIM + d] = (_Float16)v;
        } else if (mode == 2) {          // [B,H,D,T] (V transposed)
          int b = m >> 11, t = m & (T_DIM - 1);
          int h = n >> 6,  d = n & (D_DIM - 1);
          outH[(((size_t)(b * H_DIM + h) * D_DIM) + d) * T_DIM + t] = (_Float16)v;
        } else {                         // f32 + bias
          outF[(size_t)m * N + n] = v + bias[n];
        }
      }
}

// ---------------------------------------------------------------------------
// Flash attention: grid (T/64, B*H), 128 thr = 4 waves x 16 query rows.
// Q,K in [B*H,T,D]; V transposed in [B*H,D,T] so both K- and V-operand
// fragments are contiguous vector loads. Output f16 [B*T, C].
// ---------------------------------------------------------------------------
__global__ __launch_bounds__(128)
void flash_attn_kernel(const _Float16* __restrict__ Qh,
                       const _Float16* __restrict__ Kh,
                       const _Float16* __restrict__ Vt,
                       _Float16* __restrict__ Yh) {
  __shared__ __align__(16) _Float16 sP[4][16 * 64];

  int lane = threadIdx.x & 31;
  int wave = threadIdx.x >> 5;
  int bh    = blockIdx.y;
  int qTile = blockIdx.x;
  int qBase = qTile * 64 + wave * 16;

  const _Float16* Qp = Qh + (size_t)bh * T_DIM * D_DIM;
  const _Float16* Kp = Kh + (size_t)bh * T_DIM * D_DIM;
  const _Float16* Vp = Vt + (size_t)bh * D_DIM * T_DIM;   // [D][T]

  v16h qa0 = load_fragA(Qp + (size_t)qBase * D_DIM, D_DIM);
  v16h qa1 = load_fragA(Qp + (size_t)qBase * D_DIM + 32, D_DIM);

  v8f o[4] = {};
  float mrow[8], lrow[8];
#pragma unroll
  for (int r = 0; r < 8; ++r) { mrow[r] = -3.0e38f; lrow[r] = 0.0f; }

  const float scale = 0.125f;
  int rlo = (lane >> 4) << 3;
  int cIdx = lane & 15;
  _Float16* myP = sP[wave];

  for (int j = 0; j <= qTile; ++j) {
    int sBase = j * 64;

    // ---- S = Q K^T ----
    v8f s[4];
#pragma unroll
    for (int si = 0; si < 4; ++si) {
      const _Float16* Kt = Kp + (size_t)(sBase + si * 16) * D_DIM;
      v16h kb0 = load_fragB(Kt, D_DIM);
      v16h kb1 = load_fragB(Kt + 32, D_DIM);
      v8f z = {};
      z = WMMA_F16(qa0, kb0, z);
      s[si] = WMMA_F16(qa1, kb1, z);
    }

    // ---- scale + causal mask ----
#pragma unroll
    for (int si = 0; si < 4; ++si)
#pragma unroll
      for (int r = 0; r < 8; ++r) {
        float v = s[si][r] * scale;
        int row = qBase + r + rlo;
        int col = sBase + si * 16 + cIdx;
        s[si][r] = (col > row) ? -3.0e38f : v;
      }

    // ---- online softmax ----
    float mnew[8];
#pragma unroll
    for (int r = 0; r < 8; ++r) {
      float m = mrow[r];
#pragma unroll
      for (int si = 0; si < 4; ++si) m = fmaxf(m, s[si][r]);
      m = fmaxf(m, __shfl_xor(m, 1));
      m = fmaxf(m, __shfl_xor(m, 2));
      m = fmaxf(m, __shfl_xor(m, 4));
      m = fmaxf(m, __shfl_xor(m, 8));
      mnew[r] = m;
    }
#pragma unroll
    for (int r = 0; r < 8; ++r) {
      float corr = __expf(mrow[r] - mnew[r]);
      mrow[r] = mnew[r];
      lrow[r] *= corr;
#pragma unroll
      for (int d4 = 0; d4 < 4; ++d4) o[d4][r] *= corr;
    }
    float psum[8];
#pragma unroll
    for (int r = 0; r < 8; ++r) psum[r] = 0.0f;
#pragma unroll
    for (int si = 0; si < 4; ++si)
#pragma unroll
      for (int r = 0; r < 8; ++r) {
        float p = __expf(s[si][r] - mrow[r]);
        s[si][r] = p;
        psum[r] += p;
      }
#pragma unroll
    for (int r = 0; r < 8; ++r) {
      float p = psum[r];
      p += __shfl_xor(p, 1);
      p += __shfl_xor(p, 2);
      p += __shfl_xor(p, 4);
      p += __shfl_xor(p, 8);
      lrow[r] += p;
    }

    // ---- P -> LDS (C layout -> row-major 16x64), per-wave region ----
#pragma unroll
    for (int si = 0; si < 4; ++si)
#pragma unroll
      for (int r = 0; r < 8; ++r)
        myP[(size_t)(r + rlo) * 64 + si * 16 + cIdx] = (_Float16)s[si][r];
    asm volatile("s_wait_dscnt 0x0" ::: "memory");

    // ---- O += P V  (V^T layout -> contiguous B fragments) ----
    v16h pa0 = load_fragA(myP, 64);
    v16h pa1 = load_fragA(myP + 32, 64);
#pragma unroll
    for (int d4 = 0; d4 < 4; ++d4) {
      const _Float16* Vb = Vp + (size_t)(d4 * 16) * T_DIM + sBase;
      v16h vb0 = load_fragB(Vb, T_DIM);        // s = 0..31
      v16h vb1 = load_fragB(Vb + 32, T_DIM);   // s = 32..63
      o[d4] = WMMA_F16(pa0, vb0, o[d4]);
      o[d4] = WMMA_F16(pa1, vb1, o[d4]);
    }
  }

  // ---- normalize + store [B*T, C] ----
  int b = bh >> 4, h = bh & (H_DIM - 1);
#pragma unroll
  for (int d4 = 0; d4 < 4; ++d4)
#pragma unroll
    for (int r = 0; r < 8; ++r) {
      int row = qBase + r + rlo;
      size_t idx = ((size_t)(b * T_DIM + row)) * C_DIM + h * D_DIM + d4 * 16 + cIdx;
      Yh[idx] = (_Float16)(o[d4][r] / lrow[r]);
    }
}

// ---------------------------------------------------------------------------
extern "C" void kernel_launch(void* const* d_in, const int* in_sizes, int n_in,
                              void* d_out, int out_size, void* d_ws, size_t ws_size,
                              hipStream_t stream) {
  const float* X  = (const float*)d_in[0];
  const float* Wq = (const float*)d_in[1];
  const float* Wk = (const float*)d_in[2];
  const float* Wv = (const float*)d_in[3];
  const float* Wo = (const float*)d_in[4];
  const float* bo = (const float*)d_in[5];

  char* ws = (char*)d_ws;
  const size_t MB = (size_t)1 << 20;
  _Float16* Xh  = (_Float16*)(ws + 0);
  _Float16* Wqh = (_Float16*)(ws + 16 * MB);
  _Float16* Wkh = (_Float16*)(ws + 18 * MB);
  _Float16* Wvh = (_Float16*)(ws + 20 * MB);
  _Float16* Woh = (_Float16*)(ws + 22 * MB);
  _Float16* Qh  = (_Float16*)(ws + 24 * MB);   // [B*H, T, D]
  _Float16* Kh  = (_Float16*)(ws + 40 * MB);   // [B*H, T, D]
  _Float16* Vh  = (_Float16*)(ws + 56 * MB);   // [B*H, D, T]
  _Float16* Yh  = (_Float16*)(ws + 72 * MB);   // [B*T, C]

  int nX = BT_DIM * C_DIM;
  int nW = C_DIM * C_DIM;
  cvt_f32_f16_kernel<<<nX / 256, 256, 0, stream>>>(X, Xh, nX);
  cvt_f32_f16_kernel<<<nW / 256, 256, 0, stream>>>(Wq, Wqh, nW);
  cvt_f32_f16_kernel<<<nW / 256, 256, 0, stream>>>(Wk, Wkh, nW);
  cvt_f32_f16_kernel<<<nW / 256, 256, 0, stream>>>(Wv, Wvh, nW);
  cvt_f32_f16_kernel<<<nW / 256, 256, 0, stream>>>(Wo, Woh, nW);

  dim3 gg(BT_DIM / GTM, C_DIM / GTN);
  gemm_wmma_kernel<<<gg, 256, GEMM_LDS_BYTES, stream>>>(Xh, Wqh, nullptr, Qh, nullptr,
                                                        BT_DIM, C_DIM, C_DIM, 0);
  gemm_wmma_kernel<<<gg, 256, GEMM_LDS_BYTES, stream>>>(Xh, Wkh, nullptr, Kh, nullptr,
                                                        BT_DIM, C_DIM, C_DIM, 0);
  gemm_wmma_kernel<<<gg, 256, GEMM_LDS_BYTES, stream>>>(Xh, Wvh, nullptr, Vh, nullptr,
                                                        BT_DIM, C_DIM, C_DIM, 2);

  dim3 ga(T_DIM / 64, B_DIM * H_DIM);
  flash_attn_kernel<<<ga, 128, 0, stream>>>(Qh, Kh, Vh, Yh);

  gemm_wmma_kernel<<<gg, 256, GEMM_LDS_BYTES, stream>>>(Yh, Woh, bo, nullptr, (float*)d_out,
                                                        BT_DIM, C_DIM, C_DIM, 1);
}
